// SVGA_7318624272625
// MI455X (gfx1250) — compile-verified, matrix operand order checked by reading, not run on hardware
//
#include <hip/hip_runtime.h>
#include <hip/hip_bf16.h>

// ---------------------------------------------------------------------------
// GraphSAGE 2-layer encoder + unit-norm + edge BCE loss for MI455X (gfx1250).
// GEMMs via v_wmma_f32_16x16x32_f16 (wave32). Scatter via L2-resident f32
// atomics. All launches on `stream`; scratch in d_ws.
// ---------------------------------------------------------------------------

#define Nn 50000
#define Ee 800000
#define Ff 256
#define Hh 64

typedef __attribute__((ext_vector_type(16))) _Float16 v16h;
typedef __attribute__((ext_vector_type(8)))  float    v8f;
typedef __attribute__((ext_vector_type(4)))  float    v4f;

// ---------------------------------------------------------------------------
// Weight pre-swizzle: W [K x Ncols] row-major f32 -> per-lane WMMA B fragments.
// Tile (kt,nt) covers K in [kt*32, kt*32+32), N in [nt*16, nt*16+16).
// Lane l holds column n = nt*16 + (l&15); half j<8 -> K = kt*32 + (l>>4)*8 + j,
// half j>=8 -> K = kt*32 + 16 + (l>>4)*8 + (j-8). 16 contiguous f16 per lane.
// ---------------------------------------------------------------------------
__global__ void k_swz_weights(const float* __restrict__ W, _Float16* __restrict__ dst,
                              int K, int Ncols) {
    int idx = blockIdx.x * 256 + threadIdx.x;
    int total = K * Ncols;
    if (idx >= total) return;
    int ntn    = Ncols >> 4;
    int t      = idx >> 9;          // tile id (512 elems per tile)
    int within = idx & 511;
    int lane   = within >> 4;
    int j      = within & 15;
    int kt = t / ntn, nt = t % ntn;
    int ksel = (lane >> 4) * 8;
    int n = nt * 16 + (lane & 15);
    int k = kt * 32 + ((j < 8) ? (ksel + j) : (16 + ksel + (j - 8)));
    dst[idx] = (_Float16)W[k * Ncols + n];
}

__global__ void k_zero(float* __restrict__ p, long n) {
    long i = (long)blockIdx.x * blockDim.x + threadIdx.x;
    if (i < n) p[i] = 0.0f;
}

// ---------------------------------------------------------------------------
// Dual GEMM: P = A @ Bl + bias, Q = A @ Br.  A: [nrows x K] f32 row-major
// (lda == K), converted to f16 on the fly. One wave computes a 16x64 strip,
// reusing its A fragment for both B operands and all four N-tiles.
// ---------------------------------------------------------------------------
__global__ __launch_bounds__(256) void k_gemm_dual(
    const float*    __restrict__ A,
    const _Float16* __restrict__ Bl,   // swizzled [K/32][4][32][16]
    const _Float16* __restrict__ Br,
    const float*    __restrict__ bias, // [64] or nullptr
    float* __restrict__ Pout, float* __restrict__ Qout,
    int nrows, int K) {
    const int wave = threadIdx.x >> 5;
    const int lane = threadIdx.x & 31;
    const int m0 = (blockIdx.x * 8 + wave) * 16;
    if (m0 >= nrows) return;                      // wave-uniform: EXEC stays full

    const int row   = m0 + (lane & 15);
    const int ksel8 = (lane >> 4) * 8;

    v8f accP[4] = {v8f{}, v8f{}, v8f{}, v8f{}};
    v8f accQ[4] = {v8f{}, v8f{}, v8f{}, v8f{}};

    const v16h* BlT = (const v16h*)Bl;
    const v16h* BrT = (const v16h*)Br;
    const int ksteps = K >> 5;

    for (int kt = 0; kt < ksteps; ++kt) {
        // --- A fragment (ISA 16-bit 16x32 layout), f32 -> f16 convert ---
        const float* ap = A + (size_t)row * K + kt * 32 + ksel8;
        v4f f0 = *(const v4f*)(ap);
        v4f f1 = *(const v4f*)(ap + 4);
        v4f f2 = *(const v4f*)(ap + 16);
        v4f f3 = *(const v4f*)(ap + 20);
        v16h a;
#pragma unroll
        for (int i = 0; i < 4; ++i) {
            a[i]      = (_Float16)f0[i];
            a[4 + i]  = (_Float16)f1[i];
            a[8 + i]  = (_Float16)f2[i];
            a[12 + i] = (_Float16)f3[i];
        }
#pragma unroll
        for (int nt = 0; nt < 4; ++nt) {
            v16h bl = BlT[(size_t)((kt * 4 + nt) * 32 + lane)];
            v16h br = BrT[(size_t)((kt * 4 + nt) * 32 + lane)];
            accP[nt] = __builtin_amdgcn_wmma_f32_16x16x32_f16(
                false, a, false, bl, (short)0, accP[nt], false, false);
            accQ[nt] = __builtin_amdgcn_wmma_f32_16x16x32_f16(
                false, a, false, br, (short)0, accQ[nt], false, false);
        }
    }

    // --- epilogue: VGPR r -> row m0 + (lane>=16 ? 8 : 0) + r, col nt*16 + (lane&15)
    const int rowbase = m0 + (lane >> 4) * 8;
    const int col0 = lane & 15;
#pragma unroll
    for (int nt = 0; nt < 4; ++nt) {
        const int n = nt * 16 + col0;
        const float b = bias ? bias[n] : 0.0f;
#pragma unroll
        for (int r = 0; r < 8; ++r) {
            const size_t o = (size_t)(rowbase + r) * Hh + n;
            Pout[o] = accP[nt][r] + b;
            Qout[o] = accQ[nt][r];
        }
    }
}

// ---------------------------------------------------------------------------
// Edge scatter: 64 threads per edge, one f32 atomic per (edge, column).
// agg/deg fit in the 192MB L2 -> atomics resolve in L2.
// ---------------------------------------------------------------------------
__global__ void k_scatter(const int* __restrict__ src, const int* __restrict__ dst,
                          const float* __restrict__ m, float* __restrict__ agg,
                          float* __restrict__ deg, int nedges, int withDeg) {
    long t = (long)blockIdx.x * blockDim.x + threadIdx.x;
    long e = t >> 6;
    int  c = (int)(t & 63);
    if (e >= nedges) return;
    int s = src[e], d = dst[e];
    atomicAdd(&agg[(size_t)d * Hh + c], m[(size_t)s * Hh + c]);
    if (withDeg && c == 0) atomicAdd(&deg[d], 1.0f);
}

// h = relu(agg/max(deg,1) + xr)
__global__ void k_combine_relu(const float* __restrict__ agg, const float* __restrict__ deg,
                               const float* __restrict__ xr, float* __restrict__ h, long total) {
    long i = (long)blockIdx.x * blockDim.x + threadIdx.x;
    if (i >= total) return;
    float dg = fmaxf(deg[i >> 6], 1.0f);
    float v = agg[i] / dg + xr[i];
    h[i] = v > 0.0f ? v : 0.0f;
}

// z = unitnorm(agg/max(deg,1) + hr); one wave per row, 2 cols per lane.
__global__ void k_combine_norm(const float* __restrict__ agg, const float* __restrict__ deg,
                               const float* __restrict__ hr, float* __restrict__ z, int nrows) {
    int wave = threadIdx.x >> 5, lane = threadIdx.x & 31;
    int rowi = blockIdx.x * 8 + wave;
    if (rowi >= nrows) return;
    float dg = fmaxf(deg[rowi], 1.0f);
    size_t base = (size_t)rowi * Hh + lane * 2;
    float z0 = agg[base]     / dg + hr[base];
    float z1 = agg[base + 1] / dg + hr[base + 1];
    float ss = z0 * z0 + z1 * z1;
#pragma unroll
    for (int off = 16; off; off >>= 1) ss += __shfl_xor(ss, off, 32);
    if (ss == 0.0f) { z0 = 0.125f; z1 = 0.125f; ss = 1.0f; }  // degenerate row
    float r = rsqrtf(ss);
    z[base] = z0 * r;
    z[base + 1] = z1 * r;
}

// loss += softplus(-dot(z[src], z[dst])) / E ; one wave per edge.
__global__ void k_decode(const int* __restrict__ src, const int* __restrict__ dst,
                         const float* __restrict__ z, float* __restrict__ loss, int nedges) {
    int wave = threadIdx.x >> 5, lane = threadIdx.x & 31;
    long e = (long)blockIdx.x * 8 + wave;
    if (e >= nedges) return;
    int s = src[e], d = dst[e];
    const float2* zs = (const float2*)(z + (size_t)s * Hh);
    const float2* zd = (const float2*)(z + (size_t)d * Hh);
    float2 a = zs[lane], b = zd[lane];
    float p = a.x * b.x + a.y * b.y;
#pragma unroll
    for (int off = 16; off; off >>= 1) p += __shfl_xor(p, off, 32);
    if (lane == 0) {
        float sc = p;
        float v = (sc > 0.0f) ? log1pf(expf(-sc)) : (-sc + log1pf(expf(sc)));
        atomicAdd(loss, v * (1.0f / (float)nedges));
    }
}

// ---------------------------------------------------------------------------
// Launch
// ---------------------------------------------------------------------------
extern "C" void kernel_launch(void* const* d_in, const int* in_sizes, int n_in,
                              void* d_out, int out_size, void* d_ws, size_t ws_size,
                              hipStream_t stream) {
    const float* x    = (const float*)d_in[0];
    const int*   ei   = (const int*)d_in[1];
    const float* W1l  = (const float*)d_in[2];
    const float* b1l  = (const float*)d_in[3];
    const float* W1r  = (const float*)d_in[4];
    const float* W2l  = (const float*)d_in[5];
    const float* b2l  = (const float*)d_in[6];
    const float* W2r  = (const float*)d_in[7];
    const int* src = ei;
    const int* dst = ei + Ee;

    float* out  = (float*)d_out;       // z [N*H] ++ loss [1]
    float* loss = out + (size_t)Nn * Hh;

    // workspace layout (bytes, 256-aligned blocks)
    char* ws = (char*)d_ws;
    const size_t NH = (size_t)Nn * Hh * sizeof(float);     // 12.8 MB
    float* m_buf  = (float*)(ws);                          // messages (layer 1 then 2)
    float* r_buf  = (float*)(ws + NH);                     // right-branch (xr / hr)
    float* agg    = (float*)(ws + 2 * NH);                 // scatter accumulator
    float* h_buf  = (float*)(ws + 3 * NH);                 // hidden activations
    float* deg    = (float*)(ws + 4 * NH);                 // degrees [N]
    size_t woff = 4 * NH + ((size_t)Nn * sizeof(float) + 255ull & ~255ull);
    _Float16* W1l_s = (_Float16*)(ws + woff);              // 32 KB
    _Float16* W1r_s = W1l_s + (size_t)Ff * Hh;
    _Float16* W2l_s = W1r_s + (size_t)Ff * Hh;             // 8 KB each
    _Float16* W2r_s = W2l_s + (size_t)Hh * Hh;

    // 1) pre-swizzle weights into WMMA B-fragment layout
    k_swz_weights<<<(Ff * Hh + 255) / 256, 256, 0, stream>>>(W1l, W1l_s, Ff, Hh);
    k_swz_weights<<<(Ff * Hh + 255) / 256, 256, 0, stream>>>(W1r, W1r_s, Ff, Hh);
    k_swz_weights<<<(Hh * Hh + 255) / 256, 256, 0, stream>>>(W2l, W2l_s, Hh, Hh);
    k_swz_weights<<<(Hh * Hh + 255) / 256, 256, 0, stream>>>(W2r, W2r_s, Hh, Hh);

    const int gemm_blocks = (Nn + 127) / 128;   // 8 waves/block, 16 rows/wave

    // 2) layer-1 dual GEMM: m1 = x@W1l + b1l, xr = x@W1r
    k_gemm_dual<<<gemm_blocks, 256, 0, stream>>>(x, W1l_s, W1r_s, b1l,
                                                 m_buf, r_buf, Nn, Ff);

    // 3) mean-aggregate layer 1 (also computes degrees)
    k_zero<<<((long)Nn * Hh + 255) / 256, 256, 0, stream>>>(agg, (long)Nn * Hh);
    k_zero<<<(Nn + 255) / 256, 256, 0, stream>>>(deg, Nn);
    k_scatter<<<(int)(((long)Ee * 64 + 255) / 256), 256, 0, stream>>>(
        src, dst, m_buf, agg, deg, Ee, 1);

    // 4) h = relu(agg/deg + xr)
    k_combine_relu<<<((long)Nn * Hh + 255) / 256, 256, 0, stream>>>(
        agg, deg, r_buf, h_buf, (long)Nn * Hh);

    // 5) layer-2 dual GEMM: m2 = h@W2l + b2l, hr = h@W2r
    k_gemm_dual<<<gemm_blocks, 256, 0, stream>>>(h_buf, W2l_s, W2r_s, b2l,
                                                 m_buf, r_buf, Nn, Hh);

    // 6) mean-aggregate layer 2 (degrees reused)
    k_zero<<<((long)Nn * Hh + 255) / 256, 256, 0, stream>>>(agg, (long)Nn * Hh);
    k_scatter<<<(int)(((long)Ee * 64 + 255) / 256), 256, 0, stream>>>(
        src, dst, m_buf, agg, deg, Ee, 0);

    // 7) z = unitnorm(agg/deg + hr) -> d_out
    k_combine_norm<<<(Nn + 7) / 8, 256, 0, stream>>>(agg, deg, r_buf, out, Nn);

    // 8) loss = mean softplus(-<z_s, z_d>)
    k_zero<<<1, 1, 0, stream>>>(loss, 1);
    k_decode<<<Ee / 8, 256, 0, stream>>>(src, dst, out, loss, Ee);
}
